// PatchEmbedding_4355096838642
// MI455X (gfx1250) — matrix operand order repaired and checked by reading, not compiled, not run on hardware
//
#include <hip/hip_runtime.h>

// ---------------- problem constants ----------------
#define BT        96          // B*T
#define NV        390         // valid patches per (b,t)
#define MTOT      (BT * NV)   // 37440 GEMM rows
#define KTOT      672         // C*PH*PW
#define ETOT      768         // E
#define ROWLEN    312         // image width
#define CH_STRIDE 131040      // 420*312
#define BT_STRIDE 524160      // 4*420*312
#define GWIDTH    26          // grid width in patches

// ---------------- tiling ----------------
#define MTILE  128
#define NTILE  128
#define KSTEP  32
#define KSTEPS 21             // 672/32
#define LPAD   40             // halves per LDS row (80B stride -> conflict-free b128)

typedef __attribute__((ext_vector_type(4)))  float    v4f;
typedef __attribute__((ext_vector_type(8)))  float    v8f;
typedef __attribute__((ext_vector_type(4)))  _Float16 v4h;
typedef __attribute__((ext_vector_type(8)))  _Float16 v8h;
typedef __attribute__((ext_vector_type(16))) _Float16 v16h;

__global__ __launch_bounds__(256)
void patch_embed_wmma(const float* __restrict__ x,
                      const float* __restrict__ w,
                      const float* __restrict__ bias,
                      const int*   __restrict__ valid,
                      float*       __restrict__ out)
{
    __shared__ _Float16 lA[2][MTILE * LPAD];   // [m][k] tile, f16
    __shared__ _Float16 lB[2][NTILE * LPAD];   // [n][k] tile, f16 (W row-major order)

    const int tid  = threadIdx.x;
    const int lane = tid & 31;
    const int wave = tid >> 5;                 // 0..7
    const int m0   = blockIdx.y * MTILE;
    const int n0   = blockIdx.x * NTILE;

    // ---- A staging: each thread owns 4 rows {tid>>3 + 32s} at fixed k-subrun (tid&7)*4 ----
    const int arow0 = tid >> 3;                // 0..31
    const int ksub  = (tid & 7) * 4;           // 0..28, 4-aligned -> never crosses a patch row
    int  abase[4];
    bool aok[4];
#pragma unroll
    for (int s = 0; s < 4; ++s) {
        int ml = arow0 + 32 * s;
        int m  = m0 + ml;
        aok[s] = (m < MTOT);
        int mc = aok[s] ? m : (MTOT - 1);
        int bt = mc / NV;
        int vv = mc - bt * NV;
        int p  = valid[vv];
        int gy = p / GWIDTH;
        int gx = p - gy * GWIDTH;
        abase[s] = bt * BT_STRIDE + gy * (14 * ROWLEN) + gx * 12;
    }

    // ---- B staging: each thread copies 16 consecutive f32 of one W row ----
    const int brow = tid >> 1;                 // 0..127 (n within tile)
    const int bseg = (tid & 1) * 16;           // k offset within step
    const float* wrow = w + (size_t)(n0 + brow) * KTOT + bseg;

    v4f aReg[4];
    v4f bReg[4];

    auto loadAB = [&](int k0) {
        int k = k0 + ksub;
        int c = k / 168;  int r = k - c * 168;
        int i = r / 12;   int j = r - i * 12;
        int koff = c * CH_STRIDE + i * ROWLEN + j;
#pragma unroll
        for (int s = 0; s < 4; ++s) {
            v4f d = *(const v4f*)(x + abase[s] + koff);   // global_load_b128 (RT: L2 reuse)
            if (!aok[s]) d = (v4f){0.f, 0.f, 0.f, 0.f};
            aReg[s] = d;
        }
#pragma unroll
        for (int q = 0; q < 4; ++q)
            bReg[q] = *(const v4f*)(wrow + k0 + q * 4);    // L2-resident W
    };

    auto storeAB = [&](int buf) {
#pragma unroll
        for (int s = 0; s < 4; ++s) {
            v4h h;
#pragma unroll
            for (int e = 0; e < 4; ++e) h[e] = (_Float16)aReg[s][e];
            *(v4h*)&lA[buf][(arow0 + 32 * s) * LPAD + ksub] = h;   // ds_store_b64
        }
        v8h h0, h1;
#pragma unroll
        for (int e = 0; e < 4; ++e) {
            h0[e]     = (_Float16)bReg[0][e];
            h0[e + 4] = (_Float16)bReg[1][e];
            h1[e]     = (_Float16)bReg[2][e];
            h1[e + 4] = (_Float16)bReg[3][e];
        }
        *(v8h*)&lB[buf][brow * LPAD + bseg + 0] = h0;              // ds_store_b128
        *(v8h*)&lB[buf][brow * LPAD + bseg + 8] = h1;
    };

    // ---- wave tile: 2(M) x 4(N) waves, each 64x32 ----
    const int mw    = (wave >> 2) * 64;
    const int nw    = (wave & 3) * 32;
    const int l15   = lane & 15;
    const int ahalf = (lane & 16) ? 16 : 0;   // A: K 0-15 vs 16-31
    const int bhalf = (lane & 16) ? 16 : 0;   // B: K 0-15 vs 16-31

    v8f acc[4][2];
#pragma unroll
    for (int mf = 0; mf < 4; ++mf)
#pragma unroll
        for (int nf = 0; nf < 2; ++nf)
            acc[mf][nf] = (v8f){0.f,0.f,0.f,0.f,0.f,0.f,0.f,0.f};

    // prologue: stage k-step 0
    loadAB(0);
    storeAB(0);
    __syncthreads();

    for (int s = 0; s < KSTEPS; ++s) {
        const int cur = s & 1;
        if (s + 1 < KSTEPS) loadAB((s + 1) * KSTEP);   // overlap global loads with math

        v16h bf[2];
#pragma unroll
        for (int nf = 0; nf < 2; ++nf) {
            const _Float16* bp = &lB[cur][(nw + 16 * nf + l15) * LPAD + bhalf];
            v8h lo = *(const v8h*)bp;                   // ds_load_b128
            v8h hi = *(const v8h*)(bp + 8);
            bf[nf] = __builtin_shufflevector(lo, hi, 0,1,2,3,4,5,6,7,8,9,10,11,12,13,14,15);
        }
#pragma unroll
        for (int mf = 0; mf < 4; ++mf) {
            const _Float16* ap = &lA[cur][(mw + 16 * mf + l15) * LPAD + ahalf];
            v8h lo = *(const v8h*)ap;                   // ds_load_b128
            v8h hi = *(const v8h*)(ap + 8);
            v16h af = __builtin_shufflevector(lo, hi, 0,1,2,3,4,5,6,7,8,9,10,11,12,13,14,15);
#pragma unroll
            for (int nf = 0; nf < 2; ++nf)
                acc[mf][nf] = __builtin_amdgcn_wmma_f32_16x16x32_f16(
                    false, af, false, bf[nf], (short)0, acc[mf][nf], false, false);
        }

        if (s + 1 < KSTEPS) storeAB(cur ^ 1);          // fill other buffer
        __syncthreads();
    }

    // ---- epilogue: bias add + coalesced NON-TEMPORAL stores ----
    // out is 115 MB write-once; NT keeps it from evicting x (201 MB vs 192 MB L2)
    const int colbase = n0 + nw + l15;
    const float bv0 = bias[colbase];
    const float bv1 = bias[colbase + 16];
    const int rofs = (lane & 16) ? 8 : 0;    // C layout: vgpr r -> M=r (lanes 0-15) / r+8
#pragma unroll
    for (int mf = 0; mf < 4; ++mf) {
        int mrow = m0 + mw + 16 * mf + rofs;
#pragma unroll
        for (int r = 0; r < 8; ++r) {
            int m = mrow + r;
            if (m < MTOT) {
                float* o = out + (size_t)m * ETOT + colbase;
                __builtin_nontemporal_store(acc[mf][0][r] + bv0, o);
                __builtin_nontemporal_store(acc[mf][1][r] + bv1, o + 16);
            }
        }
    }
}

extern "C" void kernel_launch(void* const* d_in, const int* in_sizes, int n_in,
                              void* d_out, int out_size, void* d_ws, size_t ws_size,
                              hipStream_t stream)
{
    (void)in_sizes; (void)n_in; (void)d_ws; (void)ws_size; (void)out_size;
    const float* x     = (const float*)d_in[0];
    const float* w     = (const float*)d_in[1];
    const float* b     = (const float*)d_in[2];
    const int*   valid = (const int*)d_in[3];
    float*       out   = (float*)d_out;

    // N fastest so the 6 N-tiles sharing one A-tile run adjacently (L2 reuse of x)
    dim3 grid(ETOT / NTILE, (MTOT + MTILE - 1) / MTILE);   // (6, 293)
    patch_embed_wmma<<<grid, dim3(256), 0, stream>>>(x, w, b, valid, out);
}
